// tcn_unit_attention_46961172414623
// MI455X (gfx1250) — matrix operand order repaired and checked by reading, not compiled, not run on hardware
//
#include <hip/hip_runtime.h>

// ---- problem constants ----
#define NB   16
#define CCH  64
#define TT   300
#define TP   304     // T padded to 16
#define VJ   25
#define NHH  4
#define BTOT (NB*VJ)          // 400 sequences
#define QR   192              // qkv rows
#define RELM 599              // 2T-1
#define RELP 608              // padded to 16
#define FRAG 512              // halves per WMMA fragment tile (32 lanes x 16)

typedef _Float16 half_t;
typedef __attribute__((ext_vector_type(16))) _Float16 v16h;
typedef __attribute__((ext_vector_type(8)))  float    v8f;
typedef unsigned int u32x4 __attribute__((ext_vector_type(4)));
typedef int          i32x4 __attribute__((ext_vector_type(4)));
typedef int          i32x8 __attribute__((ext_vector_type(8)));

#define AS3 __attribute__((address_space(3)))
#define AS1 __attribute__((address_space(1)))

#ifndef __has_builtin
#define __has_builtin(x) 0
#endif
#if __has_builtin(__builtin_amdgcn_tensor_load_to_lds)
#define HAVE_TDM 1
#else
#define HAVE_TDM 0
#endif
#if __has_builtin(__builtin_amdgcn_global_load_async_to_lds_b128)
#define HAVE_ASYNC 1
#else
#define HAVE_ASYNC 0
#endif

// D = A(16x32 f16) * B(32x16 f16) + C(16x16 f32)
__device__ __forceinline__ v8f wmma16(v16h a, v16h b, v8f c) {
  return __builtin_amdgcn_wmma_f32_16x16x32_f16(false, a, false, b, (short)0, c, false, false);
}

// Fragment-order LDS read: 32B contiguous per lane -> 2x ds_load_b128
__device__ __forceinline__ v16h load_frag(const half_t* fr) {
  int lane = threadIdx.x & 31;
  return *(const v16h*)(fr + lane * 16);
}

// ---------------------------------------------------------------------------
// Kernel 1: fused data_bn + per-joint gather + qkv 1x1 conv (GEMM 192x304,K=64)
// one workgroup per sequence b = n*V + v.  Tiles staged in WMMA-fragment order.
// ---------------------------------------------------------------------------
__global__ void __launch_bounds__(256)
k_qkv(const float* __restrict__ x,
      const float* __restrict__ dbn_g, const float* __restrict__ dbn_b,
      const float* __restrict__ dbn_m, const float* __restrict__ dbn_v,
      const float* __restrict__ qkv_w, const float* __restrict__ qkvb,
      half_t* __restrict__ qkv_h)
{
  __shared__ __attribute__((aligned(32))) half_t sWf[24 * FRAG]; // A frags (mi,kc)
  __shared__ __attribute__((aligned(32))) half_t sXf[38 * FRAG]; // B frags (kc,ti)

  const int b    = blockIdx.x;
  const int nidx = b / VJ, vidx = b % VJ;
  const int tid  = threadIdx.x;

  __builtin_prefetch(qkv_w + tid * 48, 0, 0);

  // A fragments of qkv_w: frag=(mi*2+kc); element(lane,e): m=lane&15,
  // k = kc*32 + (lane>=16?16:0) + e
  for (int j = tid; j < 24 * FRAG; j += 256) {
    int low3 = j & 7, hi8 = (j >> 3) & 1, b16 = (j >> 4) & 1;
    int m = (j >> 5) & 15, kc = (j >> 9) & 1, mi = j >> 10;
    int e = b16 * 8 + low3;
    int lane2 = hi8 * 16 + m;
    int k = kc * 32 + hi8 * 16 + b16 * 8 + low3;
    sWf[(mi * 2 + kc) * FRAG + lane2 * 16 + e] = (half_t)qkv_w[(mi * 16 + m) * 64 + k];
  }

  // B fragments of normalized x: frag=(kc*19+ti); element(lane,e):
  // c = kc*32 + (lane>=16?16:0) + e,  t = ti*16 + (lane&15)
  for (int j = tid; j < 38 * FRAG; j += 256) {
    int tl = j & 15, e = (j >> 4) & 15, hi2 = (j >> 8) & 1, f = j >> 9;
    int kc = f / 19, ti = f % 19;
    int c = kc * 32 + hi2 * 16 + e;
    int t = ti * 16 + tl;
    float val = 0.f;
    if (t < TT) {
      int ch = c * VJ + vidx;                    // data_bn channel = c*V + v
      float sc = rsqrtf(dbn_v[ch] + 1e-5f) * dbn_g[ch];
      float sh = dbn_b[ch] - dbn_m[ch] * sc;
      val = x[(((size_t)nidx * CCH + c) * TT + t) * VJ + vidx] * sc + sh;
    }
    sXf[f * FRAG + (hi2 * 16 + tl) * 16 + e] = (half_t)val;
  }
  __syncthreads();

  const int wave = tid >> 5;
  const int lane = tid & 31, n = lane & 15, hi = lane >> 4;

  for (int tile = wave; tile < 12 * 19; tile += 8) {
    int mi = tile / 19, ti = tile % 19;
    v8f acc = {};
#pragma unroll
    for (int kc = 0; kc < 2; ++kc) {
      v16h a  = load_frag(sWf + (mi * 2 + kc) * FRAG);
      v16h bm = load_frag(sXf + (kc * 19 + ti) * FRAG);
      acc = wmma16(a, bm, acc);
    }
#pragma unroll
    for (int j = 0; j < 8; ++j) {
      int mg = mi * 16 + (hi ? 8 : 0) + j;
      int tg = ti * 16 + n;
      float val = acc[j] + qkvb[mg];
      if (mg < 64) val *= 0.25f;                 // q * DKH^-0.5, DKH=16
      qkv_h[((size_t)b * QR + mg) * TP + tg] = (half_t)val;
    }
  }
}

// ---------------------------------------------------------------------------
// Kernel 2: attention. One workgroup per (b, h, 16-row t-tile).
// rel GEMM (q @ key_rel^T) replaces rel_to_abs: logits[t,s] += rel[t, s-t+T-1]
// Output written in k_out's B-fragment layout (contiguous stage there).
// ---------------------------------------------------------------------------
__global__ void __launch_bounds__(128)
k_attn(const half_t* __restrict__ qkv_h,
       const float* __restrict__ key_rel,
       half_t* __restrict__ attnF)
{
  __shared__ __attribute__((aligned(32))) half_t sQf[FRAG];        // A frag (Q^T)
  __shared__ __attribute__((aligned(32))) half_t sKf[19 * FRAG];   // B frags (K)
  __shared__ __attribute__((aligned(32))) half_t sVf[10 * FRAG];   // B frags (V^T)
  __shared__ __attribute__((aligned(32))) half_t sWgtF[10 * FRAG]; // A frags (weights)
  __shared__ float  sRel[16 * RELP];
  __shared__ float  sLog[16 * TP];
  __shared__ float  sPart[4][256];
  __shared__ float  sScr[16][8];
  __shared__ float  sRowMax[16], sRowSum[16];

  const int bid = blockIdx.x;
  const int b   = bid / (NHH * 19);
  const int r   = bid % (NHH * 19);
  const int h   = r / 19;
  const int tt  = r % 19;
  const int t0  = tt * 16;
  const int tid = threadIdx.x;
  const half_t* qbase = qkv_h + (size_t)b * QR * TP;

  __builtin_prefetch(key_rel + tid * 64, 0, 0);

  // Q^T A-fragment: row m = t-row, col k = d (pad d>=16 with 0)
  for (int j = tid; j < FRAG; j += 128) {
    int e = j & 15, lane2 = j >> 4;
    int m = lane2 & 15, hi2 = lane2 >> 4;
    int d = (hi2 ? 16 : 0) + e;
    sQf[j] = (d < 16) ? qbase[(h * 16 + d) * TP + t0 + m] : (half_t)0.f;
  }
  // K B-fragments: frag st; k=d (hi half zero), n -> s = st*16 + sl
  for (int j = tid; j < 19 * FRAG; j += 128) {
    int sl = j & 15, e = (j >> 4) & 15, hi2 = (j >> 8) & 1, f = j >> 9;
    half_t val = (half_t)0.f;
    if (!hi2) val = qbase[(64 + h * 16 + e) * TP + f * 16 + sl];
    sKf[f * FRAG + (hi2 * 16 + sl) * 16 + e] = val;
  }
  // V^T B-fragments: frag kc; k -> s = kc*32+(e<8?0:16)+hi*8+(e&7), n -> dv (zero s>=300)
  for (int j = tid; j < 10 * FRAG; j += 128) {
    int e = j & 15, dv = (j >> 4) & 15, hi2 = (j >> 8) & 1, kc = j >> 9;
    int s = kc * 32 + ((e >> 3) << 4) + hi2 * 8 + (e & 7);
    half_t val = (half_t)0.f;
    if (s < TT) val = qbase[(128 + h * 16 + dv) * TP + s];
    sVf[kc * FRAG + (hi2 * 16 + dv) * 16 + e] = val;
  }
  // pre-zero weight frag 9 (covers s=288..319; exp pass fills s<304)
  for (int j = tid; j < FRAG; j += 128) sWgtF[9 * FRAG + j] = (half_t)0.f;
  __syncthreads();

  const int wave = tid >> 5;
  const int lane = tid & 31, n = lane & 15, hi = lane >> 4;
  const v16h aQ = load_frag(sQf);

  // ---- rel = Q^T @ key_rel^T : 38 m-tiles of 16 ----
  for (int mt = wave; mt < RELP / 16; mt += 4) {
    int m0 = mt * 16;
    v16h bm;
#pragma unroll
    for (int e = 0; e < 16; ++e) {
      int d = (hi ? 16 : 0) + e;
      int m = m0 + n;
      float val = 0.f;
      if (d < 16 && m < RELM) val = key_rel[m * 16 + d];
      bm[e] = (half_t)val;
    }
    v8f acc = {};
    acc = wmma16(aQ, bm, acc);
#pragma unroll
    for (int j = 0; j < 8; ++j) {
      int trow = (hi ? 8 : 0) + j;
      sRel[trow * RELP + m0 + n] = acc[j];
    }
  }
  __syncthreads();

  // ---- logits = Q^T @ K + skewed rel : 19 s-tiles ----
  for (int st = wave; st < 19; st += 4) {
    int s0 = st * 16;
    v16h bm = load_frag(sKf + st * FRAG);
    v8f acc = {};
    acc = wmma16(aQ, bm, acc);
#pragma unroll
    for (int j = 0; j < 8; ++j) {
      int trow = (hi ? 8 : 0) + j;
      int s = s0 + n, t = t0 + trow;
      float val = -1e30f;
      if (s < TT) {
        int ridx = s - t + (TT - 1);
        if (ridx < 0) ridx = 0;                 // only hit by padded t rows
        val = acc[j] + sRel[trow * RELP + ridx];
      }
      sLog[trow * TP + s] = val;
    }
  }
  __syncthreads();

  // ---- softmax over s; exp written directly into A-fragment layout (f16) ----
  {
    int row = tid & 15, part = tid >> 4;        // 8 threads/row, 38 cols each
    float mx = -1e30f;
    for (int c = part * 38; c < part * 38 + 38; ++c)
      mx = fmaxf(mx, sLog[row * TP + c]);
    sScr[row][part] = mx;
    __syncthreads();
    if (part == 0) {
      float m2 = sScr[row][0];
      for (int p = 1; p < 8; ++p) m2 = fmaxf(m2, sScr[row][p]);
      sRowMax[row] = m2;
    }
    __syncthreads();
    float rm = sRowMax[row], sm = 0.f;
    for (int c = part * 38; c < part * 38 + 38; ++c) {
      float e2 = __expf(sLog[row * TP + c] - rm);
      sm += e2;
      int kc = c >> 5, kk = c & 31;
      int e  = (((kk >> 4) & 1) << 3) | (kk & 7);
      int hi2 = (kk >> 3) & 1;
      sWgtF[kc * FRAG + (hi2 * 16 + row) * 16 + e] = (half_t)e2;
    }
    sScr[row][part] = sm;
    __syncthreads();
    if (part == 0) {
      float s2 = 0.f;
      for (int p = 0; p < 8; ++p) s2 += sScr[row][p];
      sRowSum[row] = s2;
    }
    __syncthreads();
  }

  // ---- attn = softmax(logits) @ V^T : K split 10 chunks of 32 over 4 waves
  v8f acc = {};
  for (int kc = wave; kc < 10; kc += 4) {
    v16h a  = load_frag(sWgtF + kc * FRAG);
    v16h bm = load_frag(sVf + kc * FRAG);
    acc = wmma16(a, bm, acc);
  }
#pragma unroll
  for (int j = 0; j < 8; ++j) {
    int trow = (hi ? 8 : 0) + j;
    sPart[wave][trow * 16 + n] = acc[j];
  }
  __syncthreads();

  // reduce partials, normalize, write in k_out B-fragment layout
  for (int i = tid; i < 256; i += 128) {
    int trow = i >> 4, dv = i & 15;
    float val = (sPart[0][i] + sPart[1][i] + sPart[2][i] + sPart[3][i]) / sRowSum[trow];
    int row = h * 16 + dv;                      // combined-head channel
    int kc2 = row >> 5, k32 = row & 31;
    int hi2 = k32 >> 4, el = k32 & 15;
    attnF[((size_t)b * 38 + kc2 * 19 + tt) * FRAG + (hi2 * 16 + trow) * 16 + el]
        = (half_t)val;
  }
}

// ---------------------------------------------------------------------------
// Kernel 3: out conv (64x64) + bias + skip + BN + ReLU, one WG per sequence b.
// Stages the fragment-ready attn buffer with TDM / async-LDS when available.
// ---------------------------------------------------------------------------
__global__ void __launch_bounds__(256)
k_out(const half_t* __restrict__ attnF,
      const float* __restrict__ attn_w, const float* __restrict__ attn_bias,
      const float* __restrict__ x,
      const float* __restrict__ bn_g, const float* __restrict__ bn_b,
      const float* __restrict__ bn_m, const float* __restrict__ bn_v,
      float* __restrict__ out)
{
  __shared__ __attribute__((aligned(32))) half_t sWf[8 * FRAG];   // A frags
  __shared__ __attribute__((aligned(32))) half_t sAF[38 * FRAG];  // B frags

  const int b    = blockIdx.x;
  const int nidx = b / VJ, vidx = b % VJ;
  const int tid  = threadIdx.x;
  const half_t* src = attnF + (size_t)b * 38 * FRAG;

  __builtin_prefetch(attn_w + tid * 16, 0, 0);

  // A fragments of attn_w (64x64)
  for (int j = tid; j < 8 * FRAG; j += 256) {
    int low3 = j & 7, hi8 = (j >> 3) & 1, b16 = (j >> 4) & 1;
    int m = (j >> 5) & 15, kc = (j >> 9) & 1, mi = j >> 10;
    int e = b16 * 8 + low3;
    int lane2 = hi8 * 16 + m;
    int k = kc * 32 + hi8 * 16 + b16 * 8 + low3;
    sWf[(mi * 2 + kc) * FRAG + lane2 * 16 + e] = (half_t)attn_w[(mi * 16 + m) * 64 + k];
  }

  // B fragments: contiguous 38KB copy, already fragment-ordered by k_attn
#if HAVE_TDM
  if (tid < 32) {                                // one wave issues the DMA
    unsigned lds_off = (unsigned)(size_t)(AS3 half_t*)sAF;
    unsigned long long ga = (unsigned long long)(size_t)src;
    u32x4 g0 = { 1u,                             // count=1 descriptor
                 lds_off,
                 (unsigned)(ga & 0xFFFFFFFFu),
                 (unsigned)((ga >> 32) & 0x1FFFFFFu) | (2u << 30) }; // type=2
    // 1-D tile: data_size=8B, dim0 = 38*FRAG*2/8 = 4864 elements
    i32x8 g1 = { (int)(3u << 16),        // data_size=8B
                 (int)(4864u << 16),     // tensor_dim0[15:0]
                 (int)(1u << 16),        // tensor_dim0[31:16]=0, tensor_dim1=1
                 (int)(4864u << 16),     // tile_dim0
                 0,                      // tile_dim1=0, tile_dim2=0
                 4864, 0, 0 };           // tensor_dim0_stride, dim1_stride
    i32x4 gz4 = { 0, 0, 0, 0 };
    i32x8 gz8 = { 0, 0, 0, 0, 0, 0, 0, 0 };
    __builtin_amdgcn_tensor_load_to_lds(g0, g1, gz4, gz4, gz8, 0);
#if __has_builtin(__builtin_amdgcn_s_wait_tensorcnt)
    __builtin_amdgcn_s_wait_tensorcnt((short)0);
#else
    asm volatile("s_wait_tensorcnt 0x0" ::: "memory");
#endif
  }
#elif HAVE_ASYNC
  {
    AS3 char* ldst = (AS3 char*)(AS3 half_t*)sAF;
    AS1 char* gsrc = (AS1 char*)(AS1 half_t*)(half_t*)src;
    for (int i = tid; i < (38 * FRAG * 2) / 16; i += 256)
      __builtin_amdgcn_global_load_async_to_lds_b128(
          (AS1 void*)(gsrc + i * 16), (AS3 void*)(ldst + i * 16), 0, 0);
#if __has_builtin(__builtin_amdgcn_s_wait_asynccnt)
    __builtin_amdgcn_s_wait_asynccnt(0);
#else
    asm volatile("s_wait_asynccnt 0x0" ::: "memory");
#endif
  }
#else
  for (int i = tid; i < 38 * FRAG; i += 256) sAF[i] = src[i];
#endif
  __syncthreads();

  const int wave = tid >> 5;
  const int lane = tid & 31, n = lane & 15, hi = lane >> 4;

  for (int tile = wave; tile < 4 * 19; tile += 8) {
    int mi = tile / 19, ti = tile % 19;
    v8f acc = {};
#pragma unroll
    for (int kc = 0; kc < 2; ++kc) {
      v16h a  = load_frag(sWf + (mi * 2 + kc) * FRAG);
      v16h bm = load_frag(sAF + (kc * 19 + ti) * FRAG);
      acc = wmma16(a, bm, acc);
    }
#pragma unroll
    for (int j = 0; j < 8; ++j) {
      int dv = mi * 16 + (hi ? 8 : 0) + j;
      int t  = ti * 16 + n;
      if (t < TT) {
        size_t xi = (((size_t)nidx * CCH + dv) * TT + t) * VJ + vidx;
        float val = acc[j] + attn_bias[dv] + x[xi];
        float sc  = rsqrtf(bn_v[dv] + 1e-5f) * bn_g[dv];
        val = (val - bn_m[dv]) * sc + bn_b[dv];
        out[xi] = fmaxf(val, 0.f);
      }
    }
  }
}

// ---------------------------------------------------------------------------
extern "C" void kernel_launch(void* const* d_in, const int* in_sizes, int n_in,
                              void* d_out, int out_size, void* d_ws, size_t ws_size,
                              hipStream_t stream)
{
  const float* x      = (const float*)d_in[0];
  const float* dbn_g  = (const float*)d_in[1];
  const float* dbn_b  = (const float*)d_in[2];
  const float* dbn_m  = (const float*)d_in[3];
  const float* dbn_v  = (const float*)d_in[4];
  const float* qkv_w  = (const float*)d_in[5];
  const float* qkv_b  = (const float*)d_in[6];
  const float* keyrel = (const float*)d_in[7];
  const float* attn_w = (const float*)d_in[8];
  const float* attn_b = (const float*)d_in[9];
  const float* bn_g   = (const float*)d_in[10];
  const float* bn_b   = (const float*)d_in[11];
  const float* bn_m   = (const float*)d_in[12];
  const float* bn_v   = (const float*)d_in[13];
  float* out = (float*)d_out;

  half_t* qkv_h = (half_t*)d_ws;                             // 400*192*304 f16
  half_t* attnF = qkv_h + (size_t)BTOT * QR * TP;            // 400*38*512  f16

  k_qkv<<<BTOT, 256, 0, stream>>>(x, dbn_g, dbn_b, dbn_m, dbn_v,
                                  qkv_w, qkv_b, qkv_h);
  k_attn<<<BTOT * NHH * 19, 128, 0, stream>>>(qkv_h, keyrel, attnF);
  k_out<<<BTOT, 256, 0, stream>>>(attnF, attn_w, attn_b, x,
                                  bn_g, bn_b, bn_m, bn_v, out);
}
